// ScaledDotProductAttention_74947179316221
// MI455X (gfx1250) — compile-verified, matrix operand order checked by reading, not run on hardware
//
#include <hip/hip_runtime.h>
#include <hip/hip_bf16.h>

// Outputs (concatenated flat, return order):
//   context : (8,8,512,64)  = 2,097,152 f32   -> context[b,h,q,:] = colsum_k V[b,h,k,:]
//   attn_out: (8,8,512,512) = 16,777,216 f32  -> all exactly 1.0
// (softmax is over the 3-element scale axis, so it sums to 1 per (q,k) and the
//  entire FFT/conv/score pipeline cancels out of both outputs.)

typedef __attribute__((ext_vector_type(2)))  float    v2f;
typedef __attribute__((ext_vector_type(8)))  float    v8f;
typedef __attribute__((ext_vector_type(16))) _Float16 v16h;

#ifndef __has_builtin
#define __has_builtin(x) 0
#endif

#if __has_builtin(__builtin_amdgcn_wmma_f32_16x16x4_f32)
#define USE_F32_WMMA 1
#else
#define USE_F32_WMMA 0
#endif

// ---------------------------------------------------------------------------
// Kernel 1: attn_out = 1.0f everywhere. 16.7M floats -> 4.19M float4 stores.
// This is the bandwidth-dominant part (~67 MB of global_store_b128).
// ---------------------------------------------------------------------------
__global__ __launch_bounds__(256) void fill_ones4(float4* __restrict__ out, unsigned n4) {
    unsigned i      = blockIdx.x * blockDim.x + threadIdx.x;
    unsigned stride = gridDim.x * blockDim.x;
    const float4 one = make_float4(1.0f, 1.0f, 1.0f, 1.0f);
    for (; i < n4; i += stride) out[i] = one;
}

// ---------------------------------------------------------------------------
// Kernel 2: context[b,h,q,c] = sum_k V[b,h,k,c], broadcast over q.
// One workgroup per (b,h) (64 total), 4 waves, wave w owns 16-column tile w.
// Reduction done with V_WMMA_F32_16X16X4_F32: A = ones(16x4), B = V rows
// (4x16 chunk), C accumulates. With A==1, D[m,n] = sum of everything placed
// in column n, so only the lane->column mapping of B must be correct; the
// row placement within a chunk is irrelevant to the final column sums.
// Every lane's accumulator element 0 ends up holding colsum[tile*16 + lane%16].
// EXEC is all-ones (no divergence) as WMMA requires.
// ---------------------------------------------------------------------------
__global__ __launch_bounds__(128) void colsum_bcast_wmma(const float* __restrict__ V,
                                                         float* __restrict__ ctx) {
    const int bh   = blockIdx.x;          // 0..63 == b*8 + h
    const int lane = threadIdx.x & 31;
    const int tile = threadIdx.x >> 5;    // 0..3 : 16-column tile of d=64
    const int half = lane >> 4;           // 0 | 1
    const int col  = lane & 15;           // column within tile

    const size_t base = (size_t)bh * 512 * 64;
    const float* v = V + base + (size_t)(tile * 16 + col);

    v8f acc = {};

#if USE_F32_WMMA
    v2f a;
    a[0] = 1.0f;
    a[1] = 1.0f;
    for (int k = 0; k < 512; k += 4) {
        // B 4x16 f32 chunk: VGPR0 = rows {k, k+2} (lanes 0-15 / 16-31),
        //                   VGPR1 = rows {k+1, k+3}. (Row order immaterial.)
        v2f b;
        b[0] = v[(size_t)(k + 2 * half + 0) * 64];
        b[1] = v[(size_t)(k + 2 * half + 1) * 64];
        acc = __builtin_amdgcn_wmma_f32_16x16x4_f32(
            /*neg_a=*/false, a, /*neg_b=*/false, b,
            /*c_mod=*/(short)0, acc, /*reuse_a=*/false, /*reuse_b=*/false);
    }
#else
    // Fallback: codegen-confirmed f16 WMMA; convert V rows to f16 per chunk.
    v16h a;
    #pragma unroll
    for (int j = 0; j < 16; ++j) a[j] = (_Float16)1.0f;
    for (int k = 0; k < 512; k += 32) {
        v16h b;
        #pragma unroll
        for (int j = 0; j < 16; ++j)
            b[j] = (_Float16)v[(size_t)(k + half * 16 + j) * 64];
        acc = __builtin_amdgcn_wmma_f32_16x16x32_f16(
            false, a, false, b, (short)0, acc, false, false);
    }
#endif

    const float s = acc[0];  // every D row equals the column sum

    // Broadcast to all 512 q rows; halves of the wave split the q range so the
    // two copies of each column value do independent work.
    float* c = ctx + base + (size_t)(tile * 16 + col);
    const int q0 = half * 256;
    #pragma unroll 4
    for (int q = 0; q < 256; ++q)
        c[(size_t)(q0 + q) * 64] = s;
}

extern "C" void kernel_launch(void* const* d_in, const int* in_sizes, int n_in,
                              void* d_out, int out_size, void* d_ws, size_t ws_size,
                              hipStream_t stream) {
    (void)in_sizes; (void)n_in; (void)out_size; (void)d_ws; (void)ws_size;

    // setup_inputs order: Q, K, V, attn_mask, wq1, bq1, wk1, bk1, ...
    const float* V = (const float*)d_in[2];

    float* ctx  = (float*)d_out;                       // 8*8*512*64  = 2,097,152
    float* attn = ctx + (size_t)8 * 8 * 512 * 64;      // 8*8*512*512 = 16,777,216

    const unsigned n4 = (8u * 8u * 512u * 512u) / 4u;  // 4,194,304 float4s
    fill_ones4<<<4096, 256, 0, stream>>>((float4*)attn, n4);
    colsum_bcast_wmma<<<64, 128, 0, stream>>>(V, ctx);
}
